// DSSSM_27882927685672
// MI455X (gfx1250) — compile-verified, matrix operand order checked by reading, not run on hardware
//
#include <hip/hip_runtime.h>

// ---------------------------------------------------------------------------
// Types for CDNA5 WMMA (wave32, v_wmma_f32_16x16x32_bf16)
// ---------------------------------------------------------------------------
typedef __bf16 bf16_t;
typedef __attribute__((ext_vector_type(16))) __bf16 v16bf;
typedef __attribute__((ext_vector_type(8)))  __bf16 v8bf;
typedef __attribute__((ext_vector_type(8)))  float  v8f;

union FragBf { v16bf v; v8bf h[2]; };

// Model dims
#define TT 256
#define BB 128
#define XX 128
#define YY 256
#define HH 512
#define ZZ 256
#define DD 8
#define G3H 1536   // 3*H

// ---------------------------------------------------------------------------
// Device helpers
// ---------------------------------------------------------------------------
__device__ __forceinline__ unsigned pcg(unsigned x) {
    x = x * 747796405u + 2891336453u;
    unsigned w = ((x >> ((x >> 28u) + 4u)) ^ x) * 277803737u;
    return (w >> 22u) ^ w;
}
__device__ __forceinline__ unsigned hash3(unsigned a, unsigned b, unsigned c) {
    return pcg(a * 0x85ebca6bu ^ pcg(b * 0xc2b2ae35u ^ pcg(c + 0x9e3779b9u)));
}
__device__ __forceinline__ float u01(unsigned s) {
    return ((pcg(s) >> 8) + 0.5f) * (1.0f / 16777216.0f);
}
__device__ __forceinline__ float sigmoidf_(float x) { return 1.0f / (1.0f + __expf(-x)); }
__device__ __forceinline__ float softplusf_(float x) {
    return (x > 20.0f) ? x : __logf(1.0f + __expf(x));
}

// ---------------------------------------------------------------------------
// Multi-set batched GEMM descriptor.
//   C[set][d] = act( [A0 | A1][d] (MxK) * Bt[set][d]^T (NxK) + bias[set][d] )
// A is split along K at w0 (w0 % 32 == 0); A1 may be null (no split).
// Bt row-major [N,K] == torch weight [out,in].  K must be a multiple of 64.
// ---------------------------------------------------------------------------
struct GemmSet {
    const bf16_t* A0;
    const bf16_t* A1;
    const bf16_t* Bt;
    const float*  bias;
    float*        Cf;
    bf16_t*       Cb;
    long sA0, sA1, sB, sBias, sC;
    int  lda0, lda1, w0, ldb, ldc, K, act, pad;
};

// 8 waves / block (2x4); wave tile 32x32 = 2x2 WMMA tiles; block tile 64x128.
// Ping-pong register buffers over a x2-unrolled K loop: no inter-iteration
// register copies (no v_mov shuffles / VALU WAR hazards in the hot loop).
__global__ __launch_bounds__(256) void k_gemmN(
    GemmSet s0, GemmSet s1, GemmSet s2, GemmSet s3, int E, int M, int N)
{
    const int z = blockIdx.z;
    const int setid = z / E, d = z % E;
    GemmSet S = (setid == 0) ? s0 : (setid == 1) ? s1 : (setid == 2) ? s2 : s3;

    const bf16_t* A0 = S.A0 + (long)d * S.sA0;
    const bf16_t* A1 = S.A1 ? (S.A1 + (long)d * S.sA1) : nullptr;
    const bf16_t* Bt = S.Bt + (long)d * S.sB;
    const float* bias = S.bias ? (S.bias + (long)d * S.sBias) : nullptr;
    const long coff = (long)d * S.sC;

    const int wave = threadIdx.x >> 5;     // 0..7
    const int lane = threadIdx.x & 31;
    const int wr = wave >> 2;              // 0..1
    const int wc = wave & 3;               // 0..3
    const int tileM0 = blockIdx.y * 64 + wr * 32;
    const int tileN0 = blockIdx.x * 128 + wc * 32;

    const int lm  = lane & 15;
    const int kb8 = (lane >> 4) * 8;       // K sub-offset per ISA 16-bit A/B layout

    auto loadA = [&](int k0, v16bf* a) {
#pragma unroll
        for (int i = 0; i < 2; ++i) {
            int row = tileM0 + i * 16 + lm; if (row >= M) row = M - 1;
            const bf16_t* p = (k0 < S.w0)
                ? (A0 + (long)row * S.lda0 + (k0 + kb8))
                : (A1 + (long)row * S.lda1 + (k0 - S.w0 + kb8));
            FragBf f;
            f.h[0] = *(const v8bf*)(p);
            f.h[1] = *(const v8bf*)(p + 16);
            a[i] = f.v;
        }
    };
    auto loadB = [&](int k0, v16bf* b) {
#pragma unroll
        for (int j = 0; j < 2; ++j) {
            int col = tileN0 + j * 16 + lm; if (col >= N) col = N - 1;
            const bf16_t* p = Bt + (long)col * S.ldb + (k0 + kb8);
            FragBf f;
            f.h[0] = *(const v8bf*)(p);
            f.h[1] = *(const v8bf*)(p + 16);
            b[j] = f.v;
        }
    };

    v8f acc[2][2] = {};
    auto mma = [&](v16bf* a, v16bf* b) {
#pragma unroll
        for (int i = 0; i < 2; ++i)
#pragma unroll
            for (int j = 0; j < 2; ++j)
                acc[i][j] = __builtin_amdgcn_wmma_f32_16x16x32_bf16(
                    false, a[i], false, b[j], (short)0, acc[i][j], false, false);
    };

    // K % 64 == 0 for every GEMM in this model (128/256/512/768).
    v16bf a0[2], b0[2], a1[2], b1[2];
    loadA(0, a0);
    loadB(0, b0);
    for (int k0 = 0; k0 < S.K; k0 += 64) {
        loadA(k0 + 32, a1);
        loadB(k0 + 32, b1);
        // Speculative prefetch of the weight stream one chunk further ahead
        // (gfx1250 global_prefetch; silently dropped if OOB).
        __builtin_prefetch((const void*)(Bt + (long)(tileN0 + lm) * S.ldb + k0 + 64), 0, 1);
        mma(a0, b0);
        if (k0 + 64 < S.K) {
            loadA(k0 + 64, a0);
            loadB(k0 + 64, b0);
        }
        mma(a1, b1);
    }

    // Store per 32-bit C/D layout: lane l -> col = l&15, rows = (l>>4)*8 + v
    const int mb = (lane >> 4) * 8;
#pragma unroll
    for (int i = 0; i < 2; ++i) {
#pragma unroll
        for (int j = 0; j < 2; ++j) {
            const int col = tileN0 + j * 16 + lm;
            if (col >= N) continue;
            const float bv = bias ? bias[col] : 0.0f;
#pragma unroll
            for (int v = 0; v < 8; ++v) {
                const int row = tileM0 + i * 16 + mb + v;
                if (row >= M) continue;
                float val = acc[i][j][v] + bv;
                if (S.act == 1) val = fmaxf(val, 0.0f);
                if (S.Cf) S.Cf[coff + (long)row * S.ldc + col] = val;
                if (S.Cb) S.Cb[coff + (long)row * S.ldc + col] = (bf16_t)val;
            }
        }
    }
}

// ---------------------------------------------------------------------------
// Small utility kernels
// ---------------------------------------------------------------------------
__global__ void k_cast(const float* __restrict__ s, bf16_t* __restrict__ d, long n) {
    long i = (long)blockIdx.x * blockDim.x + threadIdx.x;
    if (i < n) d[i] = (bf16_t)s[i];
}

__global__ void k_zerou32(unsigned* __restrict__ p, long n) {
    long i = (long)blockIdx.x * blockDim.x + threadIdx.x;
    if (i < n) p[i] = 0u;
}

// GRU gate fuse: r,z,n from gi (input proj + bih) and gh (hidden proj + bhh)
__global__ void k_gru_gate(const float* __restrict__ gi, const float* __restrict__ gh,
                           float* __restrict__ h, bf16_t* __restrict__ hbf,
                           bf16_t* __restrict__ out_t)
{
    int idx = blockIdx.x * blockDim.x + threadIdx.x;
    if (idx >= BB * HH) return;
    const int b = idx / HH, j = idx % HH;
    const float* gib = gi + (long)b * G3H;
    const float* ghb = gh + (long)b * G3H;
    const float r = sigmoidf_(gib[j] + ghb[j]);
    const float z = sigmoidf_(gib[HH + j] + ghb[HH + j]);
    const float n = tanhf(gib[2 * HH + j] + r * ghb[2 * HH + j]);
    const float hn = (1.0f - z) * n + z * h[idx];
    h[idx] = hn;
    const bf16_t hb = (bf16_t)hn;
    hbf[idx] = hb;
    out_t[idx] = hb;
}

// Per-t discrete posterior: softmax over experts' D-way heads, mixing,
// categorical KL, gumbel-max sample.  One block per batch element, 64 thr.
__global__ __launch_bounds__(64) void k_dpost(
    const bf16_t* __restrict__ ob_t, const bf16_t* __restrict__ Wdp,
    const float* __restrict__ bdp, float* __restrict__ d_prev,
    float* __restrict__ d_post, const float* __restrict__ logT,
    float* __restrict__ out, int t)
{
    __shared__ float pa[64];
    __shared__ float red[64];
    __shared__ float dpnew[8];
    __shared__ int best;
    const int b = blockIdx.x;
    const int tid = threadIdx.x, d = tid >> 3, i = tid & 7;

    float s = bdp[d * 8 + i];
    const bf16_t* w = Wdp + (long)(d * 8 + i) * HH;
    const bf16_t* o = ob_t + (long)b * HH;
    for (int k = 0; k < HH; ++k) s += (float)w[k] * (float)o[k];
    pa[tid] = s;
    __syncthreads();
    float mx = -1e30f;
    for (int j = 0; j < 8; ++j) mx = fmaxf(mx, pa[d * 8 + j]);
    float den = 0.0f;
    for (int j = 0; j < 8; ++j) den += __expf(pa[d * 8 + j] - mx);
    const float p = __expf(s - mx) / den;
    __syncthreads();
    pa[tid] = p;
    __syncthreads();

    // kldc weighted by previous step's d_post (still stored in d_post)
    red[tid] = p * (__logf(p + 1e-8f) - logT[d * 8 + i]) * d_post[b * 8 + d];
    __syncthreads();
    for (int st = 32; st > 0; st >>= 1) {
        if (tid < st) red[tid] += red[tid + st];
        __syncthreads();
    }
    if (tid == 0) atomicAdd(out + 1, red[0]);

    if (tid < 8) {
        float dp = 0.0f;
        for (int dd = 0; dd < 8; ++dd) dp += pa[dd * 8 + tid] * d_prev[b * 8 + dd];
        dpnew[tid] = dp;
    }
    __syncthreads();
    if (tid == 0) {
        float bv = -1e30f; int bi = 0;
        for (int j = 0; j < 8; ++j) {
            const float u = u01(hash3((unsigned)t, (unsigned)b, (unsigned)j));
            const float g = -__logf(-__logf(u));
            const float v = __logf(dpnew[j] + 1e-20f) + g;
            if (v > bv) { bv = v; bi = j; }
        }
        best = bi;
    }
    __syncthreads();
    if (tid < 8) {
        d_post[b * 8 + tid] = dpnew[tid];
        d_prev[b * 8 + tid] = (tid == best) ? 1.0f : 0.0f;   // becomes d_new
    }
}

// Mix q-experts by d_new, sample z_t, accumulate gaussian KL weighted by d_post
__global__ __launch_bounds__(256) void k_mix(
    const float* __restrict__ zpm, const float* __restrict__ zps,
    const float* __restrict__ zqm, const float* __restrict__ zqs,
    const float* __restrict__ d_new, const float* __restrict__ d_post,
    bf16_t* __restrict__ zbuf, float* __restrict__ out, int t)
{
    __shared__ float red[256];
    const int idx = blockIdx.x * 256 + threadIdx.x;
    float kacc = 0.0f;
    if (idx < BB * ZZ) {
        const int b = idx / ZZ;
        float mm = 0.0f, ms = 0.0f;
        for (int d = 0; d < 8; ++d) {
            const long o = (long)d * BB * ZZ + idx;
            const float pm = zpm[o], qm = zqm[o];
            const float ps = softplusf_(zps[o]);
            const float qs = softplusf_(zqs[o]);
            const float dn = d_new[b * 8 + d];
            const float dp = d_post[b * 8 + d];
            mm += dn * qm;
            ms += dn * qs;
            const float dm = qm - pm;
            const float r = 2.0f * __logf(ps / qs) + (qs * qs + dm * dm) / (ps * ps) - 1.0f;
            kacc += 0.5f * r * dp;
        }
        const float u1 = u01(hash3((unsigned)t, (unsigned)idx, 0x1234u));
        const float u2 = u01(hash3((unsigned)t, (unsigned)idx, 0x5678u));
        const float nrm = sqrtf(-2.0f * __logf(u1)) * __cosf(6.2831853f * u2);
        zbuf[idx] = (bf16_t)(mm + ms * nrm);
    }
    red[threadIdx.x] = kacc;
    __syncthreads();
    for (int st = 128; st > 0; st >>= 1) {
        if (threadIdx.x < st) red[threadIdx.x] += red[threadIdx.x + st];
        __syncthreads();
    }
    if (threadIdx.x == 0) atomicAdd(out + 0, red[0]);
}

// Emission NLL weighted by d_post
__global__ __launch_bounds__(256) void k_nll(
    const float* __restrict__ yem, const float* __restrict__ yes_,
    const float* __restrict__ y_t, const float* __restrict__ d_post,
    float* __restrict__ out)
{
    __shared__ float red[256];
    const int idx = blockIdx.x * 256 + threadIdx.x;
    float acc = 0.0f;
    if (idx < BB * YY) {
        const int b = idx / YY;
        const float yt = y_t[idx];
        for (int d = 0; d < 8; ++d) {
            const long o = (long)d * BB * YY + idx;
            const float std = softplusf_(yes_[o]);
            const float diff = (yt - yem[o]) / std;
            acc += d_post[b * 8 + d] *
                   0.5f * (1.8378770664093453f + 2.0f * __logf(std) + diff * diff);
        }
    }
    red[threadIdx.x] = acc;
    __syncthreads();
    for (int st = 128; st > 0; st >>= 1) {
        if (threadIdx.x < st) red[threadIdx.x] += red[threadIdx.x + st];
        __syncthreads();
    }
    if (threadIdx.x == 0) atomicAdd(out + 2, red[0]);
}

// Transition prior logT = log(0.5*softmax(T0 @ Wd^T + bd) + 0.5*I), D=8
__global__ __launch_bounds__(64) void k_trans(const float* __restrict__ Wd,
                                              const float* __restrict__ bd,
                                              float* __restrict__ logT)
{
    __shared__ float lg[64];
    const int tid = threadIdx.x, i = tid >> 3, j = tid & 7;
    float s = bd[j];
    for (int k = 0; k < 8; ++k) {
        const float t0 = (i == k) ? 0.65f : 0.05f;  // eye*(1-0.4)+0.05
        s += t0 * Wd[j * 8 + k];
    }
    lg[tid] = s;
    __syncthreads();
    float mx = -1e30f;
    for (int jj = 0; jj < 8; ++jj) mx = fmaxf(mx, lg[i * 8 + jj]);
    float den = 0.0f;
    for (int jj = 0; jj < 8; ++jj) den += __expf(lg[i * 8 + jj] - mx);
    const float p = __expf(s - mx) / den;
    const float tr = 0.5f * p + ((i == j) ? 0.5f : 0.0f);
    logT[tid] = __logf(tr);
}

// Init: zero losses, z0=0, dpost0=1/8, d0 = one-hot(categorical(uniform))
__global__ void k_init(float* __restrict__ out, float* __restrict__ dprev,
                       float* __restrict__ dpost, bf16_t* __restrict__ zbuf)
{
    const int idx = blockIdx.x * blockDim.x + threadIdx.x;
    if (idx < BB * ZZ) zbuf[idx] = (bf16_t)0.0f;
    if (idx < BB * 8) {
        const int b = idx >> 3, i = idx & 7;
        const unsigned c = pcg(0xD00Du ^ (unsigned)b) & 7u;
        dpost[idx] = 0.125f;
        dprev[idx] = (i == (int)c) ? 1.0f : 0.0f;
    }
    if (idx < 3) out[idx] = 0.0f;
}

// ---------------------------------------------------------------------------
// Host orchestration
// ---------------------------------------------------------------------------
static inline size_t alignUp(size_t v, size_t a) { return (v + a - 1) / a * a; }

static inline GemmSet mkset(const bf16_t* A0, long sA0, int lda0,
                            const bf16_t* A1, long sA1, int lda1, int w0,
                            const bf16_t* Bt, long sB, int ldb,
                            const float* bias, long sBias,
                            float* Cf, bf16_t* Cb, long sC, int ldc,
                            int K, int act)
{
    GemmSet s;
    s.A0 = A0; s.A1 = A1; s.Bt = Bt; s.bias = bias; s.Cf = Cf; s.Cb = Cb;
    s.sA0 = sA0; s.sA1 = sA1; s.sB = sB; s.sBias = sBias; s.sC = sC;
    s.lda0 = lda0; s.lda1 = lda1; s.w0 = w0; s.ldb = ldb; s.ldc = ldc;
    s.K = K; s.act = act; s.pad = 0;
    return s;
}

extern "C" void kernel_launch(void* const* d_in, const int* in_sizes, int n_in,
                              void* d_out, int out_size, void* d_ws, size_t ws_size,
                              hipStream_t stream)
{
    (void)in_sizes; (void)n_in; (void)out_size; (void)ws_size;

    const float* x_f   = (const float*)d_in[0];
    const float* y_f   = (const float*)d_in[1];
    const float* Wdpr  = (const float*)d_in[2];
    const float* bdpr  = (const float*)d_in[3];
    const float* Wdp_f = (const float*)d_in[4];
    const float* bdp   = (const float*)d_in[5];
    const float* Wq1_f = (const float*)d_in[6];   const float* bq1 = (const float*)d_in[7];
    const float* Wq2_f = (const float*)d_in[8];   const float* bq2 = (const float*)d_in[9];
    const float* Wqm_f = (const float*)d_in[10];  const float* bqm = (const float*)d_in[11];
    const float* Wqs_f = (const float*)d_in[12];  const float* bqs = (const float*)d_in[13];
    const float* Wp1_f = (const float*)d_in[14];  const float* bp1 = (const float*)d_in[15];
    const float* Wp2_f = (const float*)d_in[16];  const float* bp2 = (const float*)d_in[17];
    const float* Wpm_f = (const float*)d_in[18];  const float* bpm = (const float*)d_in[19];
    const float* Wps_f = (const float*)d_in[20];  const float* bps = (const float*)d_in[21];
    const float* We1_f = (const float*)d_in[22];  const float* be1 = (const float*)d_in[23];
    const float* We2_f = (const float*)d_in[24];  const float* be2 = (const float*)d_in[25];
    const float* Wem_f = (const float*)d_in[26];  const float* bem = (const float*)d_in[27];
    const float* Wes_f = (const float*)d_in[28];  const float* bes = (const float*)d_in[29];
    const float* Wihf_f = (const float*)d_in[30]; const float* Whhf_f = (const float*)d_in[31];
    const float* bihf  = (const float*)d_in[32];  const float* bhhf  = (const float*)d_in[33];
    const float* Wihb_f = (const float*)d_in[34]; const float* Whhb_f = (const float*)d_in[35];
    const float* bihb  = (const float*)d_in[36];  const float* bhhb  = (const float*)d_in[37];

    float* out = (float*)d_out;

    // ---- workspace bump allocator ----
    char* ws = (char*)d_ws;
    size_t off = 0;
    auto alloc = [&](size_t bytes) -> void* {
        void* p = ws + off;
        off = alignUp(off + bytes, 256);
        return p;
    };
    auto castW = [&](const float* src, size_t n) -> bf16_t* {
        bf16_t* p = (bf16_t*)alloc(n * sizeof(bf16_t));
        const long nb = (long)((n + 255) / 256);
        k_cast<<<dim3((unsigned)nb), 256, 0, stream>>>(src, p, (long)n);
        return p;
    };

    // bf16 weights/inputs (L2-resident working set)
    bf16_t* xb    = castW(x_f,   (size_t)TT * BB * XX);
    bf16_t* yb    = castW(y_f,   (size_t)TT * BB * YY);
    bf16_t* Wihf  = castW(Wihf_f, (size_t)G3H * XX);
    bf16_t* Whhf  = castW(Whhf_f, (size_t)G3H * HH);
    bf16_t* Wihb  = castW(Wihb_f, (size_t)G3H * (YY + HH));
    bf16_t* Whhb  = castW(Whhb_f, (size_t)G3H * HH);
    bf16_t* Wdpb  = castW(Wdp_f,  (size_t)DD * DD * HH);
    bf16_t* Wq1b  = castW(Wq1_f, (size_t)DD * ZZ * (ZZ + HH));
    bf16_t* Wq2b  = castW(Wq2_f, (size_t)DD * ZZ * ZZ);
    bf16_t* Wqmb  = castW(Wqm_f, (size_t)DD * ZZ * ZZ);
    bf16_t* Wqsb  = castW(Wqs_f, (size_t)DD * ZZ * ZZ);
    bf16_t* Wp1b  = castW(Wp1_f, (size_t)DD * ZZ * (ZZ + HH));
    bf16_t* Wp2b  = castW(Wp2_f, (size_t)DD * ZZ * ZZ);
    bf16_t* Wpmb  = castW(Wpm_f, (size_t)DD * ZZ * ZZ);
    bf16_t* Wpsb  = castW(Wps_f, (size_t)DD * ZZ * ZZ);
    bf16_t* We1b  = castW(We1_f, (size_t)DD * YY * (ZZ + HH));
    bf16_t* We2b  = castW(We2_f, (size_t)DD * YY * YY);
    bf16_t* Wemb  = castW(Wem_f, (size_t)DD * YY * YY);
    bf16_t* Wesb  = castW(Wes_f, (size_t)DD * YY * YY);

    bf16_t* ofb = (bf16_t*)alloc((size_t)TT * BB * HH * 2);
    bf16_t* obb = (bf16_t*)alloc((size_t)TT * BB * HH * 2);

    float*  gi   = (float*)alloc((size_t)BB * G3H * 4);
    float*  gh   = (float*)alloc((size_t)BB * G3H * 4);
    float*  hf   = (float*)alloc((size_t)BB * HH * 4);
    bf16_t* hb   = (bf16_t*)alloc((size_t)BB * HH * 2);
    bf16_t* hp   = (bf16_t*)alloc((size_t)DD * BB * ZZ * 2);
    bf16_t* hp2  = (bf16_t*)alloc((size_t)DD * BB * ZZ * 2);
    bf16_t* hq   = (bf16_t*)alloc((size_t)DD * BB * ZZ * 2);
    bf16_t* hq2  = (bf16_t*)alloc((size_t)DD * BB * ZZ * 2);
    bf16_t* he   = (bf16_t*)alloc((size_t)DD * BB * YY * 2);
    bf16_t* he2  = (bf16_t*)alloc((size_t)DD * BB * YY * 2);
    float*  zpm  = (float*)alloc((size_t)DD * BB * ZZ * 4);
    float*  zps  = (float*)alloc((size_t)DD * BB * ZZ * 4);
    float*  zqm  = (float*)alloc((size_t)DD * BB * ZZ * 4);
    float*  zqs  = (float*)alloc((size_t)DD * BB * ZZ * 4);
    float*  yem  = (float*)alloc((size_t)DD * BB * YY * 4);
    float*  yes_ = (float*)alloc((size_t)DD * BB * YY * 4);
    bf16_t* zbuf = (bf16_t*)alloc((size_t)BB * ZZ * 2);
    float*  dprev = (float*)alloc((size_t)BB * DD * 4);
    float*  dpost = (float*)alloc((size_t)BB * DD * 4);
    float*  logT  = (float*)alloc(64 * 4);

    GemmSet zset = {};
    auto gemmN = [&](const GemmSet* s, int nsets, int E, int M, int N) {
        dim3 grid((unsigned)((N + 127) / 128), (unsigned)((M + 63) / 64),
                  (unsigned)(nsets * E));
        k_gemmN<<<grid, 256, 0, stream>>>(s[0],
                                          nsets > 1 ? s[1] : zset,
                                          nsets > 2 ? s[2] : zset,
                                          nsets > 3 ? s[3] : zset, E, M, N);
    };
    auto zerof = [&](void* p, size_t bytes) {
        const long nw = (long)(bytes / 4);
        k_zerou32<<<dim3((unsigned)((nw + 255) / 256)), 256, 0, stream>>>((unsigned*)p, nw);
    };

    // ---- prologue ----
    k_trans<<<1, 64, 0, stream>>>(Wdpr, bdpr, logT);
    k_init<<<dim3((BB * ZZ + 255) / 256), 256, 0, stream>>>(out, dprev, dpost, zbuf);

    const int gateBlocks = (BB * HH + 255) / 256;

    // ---- forward GRU over x:  gi = x_t@Wih^T+bih, gh = h@Whh^T+bhh (one launch)
    zerof(hf, (size_t)BB * HH * 4);
    zerof(hb, (size_t)BB * HH * 2);
    for (int t = 0; t < TT; ++t) {
        GemmSet s[2];
        s[0] = mkset(xb + (size_t)t * BB * XX, 0, XX, nullptr, 0, 0, XX,
                     Wihf, 0, XX, bihf, 0, gi, nullptr, 0, G3H, XX, 0);
        s[1] = mkset(hb, 0, HH, nullptr, 0, 0, HH,
                     Whhf, 0, HH, bhhf, 0, gh, nullptr, 0, G3H, HH, 0);
        gemmN(s, 2, 1, BB, G3H);
        k_gru_gate<<<gateBlocks, 256, 0, stream>>>(gi, gh, hf, hb,
                                                   ofb + (size_t)t * BB * HH);
    }

    // ---- backward GRU over flip(concat(y, of)):  split-A removes the concat
    zerof(hf, (size_t)BB * HH * 4);
    zerof(hb, (size_t)BB * HH * 2);
    for (int tp = 0; tp < TT; ++tp) {
        const int tr = TT - 1 - tp;
        GemmSet s[2];
        s[0] = mkset(yb + (size_t)tr * BB * YY, 0, YY,
                     ofb + (size_t)tr * BB * HH, 0, HH, YY,
                     Wihb, 0, YY + HH, bihb, 0, gi, nullptr, 0, G3H, YY + HH, 0);
        s[1] = mkset(hb, 0, HH, nullptr, 0, 0, HH,
                     Whhb, 0, HH, bhhb, 0, gh, nullptr, 0, G3H, HH, 0);
        gemmN(s, 2, 1, BB, G3H);
        k_gru_gate<<<gateBlocks, 256, 0, stream>>>(gi, gh, hf, hb,
                                                   obb + (size_t)tr * BB * HH);
    }

    // ---- sequential step loop (9 launches / step) ----
    const long sW1 = (long)ZZ * (ZZ + HH);   // expert weight strides
    const long sW2 = (long)ZZ * ZZ;
    const long sE1 = (long)YY * (ZZ + HH);
    const long sE2 = (long)YY * YY;
    const long sCz = (long)BB * ZZ;
    const long sCy = (long)BB * YY;
    const int  mixBlocks = (BB * ZZ + 255) / 256;
    const int  nllBlocks = (BB * YY + 255) / 256;

    for (int t = 0; t < TT; ++t) {
        const bf16_t* of_t = ofb + (size_t)t * BB * HH;
        const bf16_t* ob_t = obb + (size_t)t * BB * HH;

        k_dpost<<<BB, 64, 0, stream>>>(ob_t, Wdpb, bdp, dprev, dpost, logT, out, t);

        // Layer-1 of p-net and q-net fused; A = [of|z_prev] / [ob|z_prev]
        {
            GemmSet s[2];
            s[0] = mkset(of_t, 0, HH, zbuf, 0, ZZ, HH,
                         Wp1b, sW1, ZZ + HH, bp1, ZZ, nullptr, hp, sCz, ZZ, ZZ + HH, 1);
            s[1] = mkset(ob_t, 0, HH, zbuf, 0, ZZ, HH,
                         Wq1b, sW1, ZZ + HH, bq1, ZZ, nullptr, hq, sCz, ZZ, ZZ + HH, 1);
            gemmN(s, 2, DD, BB, ZZ);
        }
        // Layer-2 of p-net and q-net fused
        {
            GemmSet s[2];
            s[0] = mkset(hp, sCz, ZZ, nullptr, 0, 0, ZZ,
                         Wp2b, sW2, ZZ, bp2, ZZ, nullptr, hp2, sCz, ZZ, ZZ, 1);
            s[1] = mkset(hq, sCz, ZZ, nullptr, 0, 0, ZZ,
                         Wq2b, sW2, ZZ, bq2, ZZ, nullptr, hq2, sCz, ZZ, ZZ, 1);
            gemmN(s, 2, DD, BB, ZZ);
        }
        // All four z-heads fused (mean/std for p and q)
        {
            GemmSet s[4];
            s[0] = mkset(hp2, sCz, ZZ, nullptr, 0, 0, ZZ,
                         Wpmb, sW2, ZZ, bpm, ZZ, zpm, nullptr, sCz, ZZ, ZZ, 0);
            s[1] = mkset(hp2, sCz, ZZ, nullptr, 0, 0, ZZ,
                         Wpsb, sW2, ZZ, bps, ZZ, zps, nullptr, sCz, ZZ, ZZ, 0);
            s[2] = mkset(hq2, sCz, ZZ, nullptr, 0, 0, ZZ,
                         Wqmb, sW2, ZZ, bqm, ZZ, zqm, nullptr, sCz, ZZ, ZZ, 0);
            s[3] = mkset(hq2, sCz, ZZ, nullptr, 0, 0, ZZ,
                         Wqsb, sW2, ZZ, bqs, ZZ, zqs, nullptr, sCz, ZZ, ZZ, 0);
            gemmN(s, 4, DD, BB, ZZ);
        }

        // mix + sample z_t (writes zbuf), accumulate gaussian KL
        k_mix<<<mixBlocks, 256, 0, stream>>>(zpm, zps, zqm, zqs, dprev, dpost, zbuf, out, t);

        // emission experts; A = [of | z_t] via split-A
        {
            GemmSet s[1];
            s[0] = mkset(of_t, 0, HH, zbuf, 0, ZZ, HH,
                         We1b, sE1, ZZ + HH, be1, YY, nullptr, he, sCy, YY, ZZ + HH, 1);
            gemmN(s, 1, DD, BB, YY);
        }
        {
            GemmSet s[1];
            s[0] = mkset(he, sCy, YY, nullptr, 0, 0, YY,
                         We2b, sE2, YY, be2, YY, nullptr, he2, sCy, YY, YY, 1);
            gemmN(s, 1, DD, BB, YY);
        }
        // Emission heads fused
        {
            GemmSet s[2];
            s[0] = mkset(he2, sCy, YY, nullptr, 0, 0, YY,
                         Wemb, sE2, YY, bem, YY, yem, nullptr, sCy, YY, YY, 0);
            s[1] = mkset(he2, sCy, YY, nullptr, 0, 0, YY,
                         Wesb, sE2, YY, bes, YY, yes_, nullptr, sCy, YY, YY, 0);
            gemmN(s, 2, DD, BB, YY);
        }

        k_nll<<<nllBlocks, 256, 0, stream>>>(yem, yes_, y_f + (size_t)t * BB * YY, dpost, out);
    }
}
